// SkipGram_53171695124812
// MI455X (gfx1250) — matrix operand order repaired and checked by reading, not compiled, not run on hardware
//
#include <hip/hip_runtime.h>

// ---------------------------------------------------------------------------
// SkipGram double-softmax NLL, fused gather + bf16-WMMA GEMM + two-pass
// online softmax.  Target: gfx1250 (CDNA5, wave32, WMMA).
// Round 4: 32-row M-tile per workgroup (2 A-tiles per wave) -> every B
// fragment feeds 8 WMMAs in two independent accumulator chains, halving the
// per-pass L2 traffic for W_dec and doubling FLOP/byte.  Keeps the
// 2x-unrolled ping-pong load pipeline (fixed buffer roles, clamped indices).
// ---------------------------------------------------------------------------

typedef __attribute__((ext_vector_type(16))) __bf16 v16bf;
typedef __attribute__((ext_vector_type(8)))  float  v8f;

#define NUM_NODES 50000
#define ENC_DIM   128
#define BATCH     256
#define WALK_LEN  20
#define T_TOTAL   (BATCH * WALK_LEN)   // 5120 rows
#define MTILES    2                    // 16-row WMMA tiles per workgroup
#define ROWS      (16 * MTILES)        // 32 rows per workgroup
#define NWAVES    8
#define NTHREADS  (NWAVES * 32)
#define EMB_LD    132                  // padded LDS stride (132 % 64 = 4 -> no bank conflicts)
#define NCHUNK    (NUM_NODES / 16)     // 3125 column chunks of 16

__global__ void zero_out_kernel(float* out) { out[0] = 0.0f; }

__global__ void cvt_wdec_kernel(const float* __restrict__ src,
                                __bf16* __restrict__ dst, int n) {
  int i = blockIdx.x * blockDim.x + threadIdx.x;
  int stride = gridDim.x * blockDim.x;
  for (; i < n; i += stride) dst[i] = (__bf16)src[i];
}

// One chunk's worth of B fragments: 4 K-steps of a 32x16 bf16 tile.
struct BFrag { v16bf b[4]; };

// B 32x16 bf16, contiguous k order chosen here (A matches): lanes 0-15 hold K=0..15, lanes 16-31 hold K=16..31,
// column n = lane%16.  W_dec rows are K-contiguous, so each lane does two
// 16B vector loads per K-step.
template <bool PRECVT>
__device__ __forceinline__ BFrag load_chunk(const float* __restrict__ Wf,
                                            const __bf16* __restrict__ Wb,
                                            int c, int lmod, int half) {
  const size_t off = (size_t)(c * 16 + lmod) * ENC_DIM;
  BFrag f;
#pragma unroll
  for (int ks = 0; ks < 4; ++ks) {
    const int kb = ks * 32 + half * 16;
    if (PRECVT) {
      union { v16bf v; uint4 q[2]; } u;
      u.q[0] = *(const uint4*)(Wb + off + kb);
      u.q[1] = *(const uint4*)(Wb + off + kb + 8);
      f.b[ks] = u.v;
    } else {
      const float* p = Wf + off + kb;
#pragma unroll
      for (int i = 0; i < 16; ++i) f.b[ks][i] = (__bf16)p[i];
    }
  }
  return f;
}

// Two 16x16 output tiles sharing one B chunk: 8 WMMAs, 2 independent chains.
__device__ __forceinline__ void tile_wmma2(const v16bf (&af)[MTILES][4],
                                           const BFrag& bf, v8f& a0, v8f& a1) {
  a0 = (v8f){};
  a1 = (v8f){};
#pragma unroll
  for (int ks = 0; ks < 4; ++ks) {
    a0 = __builtin_amdgcn_wmma_f32_16x16x32_bf16(
        false, af[0][ks], false, bf.b[ks], (short)0, a0, false, false);
    a1 = __builtin_amdgcn_wmma_f32_16x16x32_bf16(
        false, af[1][ks], false, bf.b[ks], (short)0, a1, false, false);
  }
}

template <bool PRECVT>
__launch_bounds__(NTHREADS)
__global__ void skipgram_loss_kernel(const float* __restrict__ W_enc,
                                     const float* __restrict__ b_enc,
                                     const float* __restrict__ W_dec,
                                     const __bf16* __restrict__ W_dec_bf,
                                     const float* __restrict__ b_dec,
                                     const int* __restrict__ target,
                                     const int* __restrict__ rw,
                                     float* __restrict__ out) {
  __shared__ float emb[ROWS * EMB_LD];        // f32 emb tile (32 x 128)
  __shared__ float wred_m[NWAVES][ROWS];      // per-wave softmax partials
  __shared__ float wred_s[NWAVES][ROWS];
  __shared__ float M1[ROWS], S1[ROWS], PINV[ROWS], TLOGIT[ROWS];

  const int tid  = threadIdx.x;
  const int lane = tid & 31;
  const int wave = tid >> 5;
  const int half = lane >> 4;
  const int lmod = lane & 15;
  const int t0   = blockIdx.x * ROWS;

  // ---- Stage emb tile: emb[r][e] = W_enc[e][rw[t0+r]] + b_enc[e] -----------
  for (int i = tid; i < ROWS * ENC_DIM; i += NTHREADS) {
    const int r = i >> 7, e = i & 127;
    const int node = rw[t0 + r];
    emb[r * EMB_LD + e] = W_enc[(size_t)e * NUM_NODES + node] + b_enc[e];
  }
  if (tid < ROWS) {
    const int node = target[(t0 + tid) / WALK_LEN];
    TLOGIT[tid] = b_dec[node];
  }
  __syncthreads();

  // ---- Target-column logit: logit_tgt[r] = emb[r] . W_dec[tgt[r]] + b_dec --
  {
    const int r = tid >> 3, seg = tid & 7;     // 32 rows x 8 segments of 16
    const int node = target[(t0 + r) / WALK_LEN];
    const float* wr = W_dec + (size_t)node * ENC_DIM + seg * 16;
    const float* er = emb + r * EMB_LD + seg * 16;
    float p = 0.0f;
#pragma unroll
    for (int j = 0; j < 16; ++j) p += er[j] * wr[j];
    atomicAdd(&TLOGIT[r], p);
  }

  // ---- Build bf16 A fragments (constant for the whole kernel) --------------
  // A 16x32 bf16, contiguous k order chosen here (B matches): lane (half,m): elems 0-7 -> K = half*16 + i,
  //                                     elems 8-15 -> K = half*16 + 8 + i
  v16bf afrag[MTILES][4];
#pragma unroll
  for (int mt = 0; mt < MTILES; ++mt) {
    const float* er = emb + (mt * 16 + lmod) * EMB_LD;
#pragma unroll
    for (int ks = 0; ks < 4; ++ks) {
      const int kb = ks * 32 + half * 16;
#pragma unroll
      for (int i = 0; i < 8; ++i) afrag[mt][ks][i]     = (__bf16)er[kb + i];
#pragma unroll
      for (int i = 0; i < 8; ++i) afrag[mt][ks][8 + i] = (__bf16)er[kb + 8 + i];
    }
  }

  // =========================== PASS 1: (M1, S1) ============================
  float m_run[MTILES][8], s_run[MTILES][8];
#pragma unroll
  for (int mt = 0; mt < MTILES; ++mt)
#pragma unroll
    for (int v = 0; v < 8; ++v) { m_run[mt][v] = -3.0e38f; s_run[mt][v] = 0.0f; }

  auto upd1 = [&](const v8f& a0, const v8f& a1, float bd) {
#pragma unroll
    for (int v = 0; v < 8; ++v) {               // row = mt*16 + v + 8*half
      {
        const float lg = a0[v] + bd;
        const float mo = m_run[0][v];
        const float mn = fmaxf(mo, lg);
        s_run[0][v] = s_run[0][v] * __expf(mo - mn) + __expf(lg - mn);
        m_run[0][v] = mn;
      }
      {
        const float lg = a1[v] + bd;
        const float mo = m_run[1][v];
        const float mn = fmaxf(mo, lg);
        s_run[1][v] = s_run[1][v] * __expf(mo - mn) + __expf(lg - mn);
        m_run[1][v] = mn;
      }
    }
  };

  {
    int c = wave;
    BFrag b0 = load_chunk<PRECVT>(W_dec, W_dec_bf, c, lmod, half);
    v8f a0, a1;
    for (; c + NWAVES < NCHUNK; c += 2 * NWAVES) {
      BFrag b1 = load_chunk<PRECVT>(W_dec, W_dec_bf, c + NWAVES, lmod, half);
      tile_wmma2(afrag, b0, a0, a1);
      upd1(a0, a1, b_dec[c * 16 + lmod]);
      const int cn = (c + 2 * NWAVES < NCHUNK) ? (c + 2 * NWAVES) : (NCHUNK - 1);
      b0 = load_chunk<PRECVT>(W_dec, W_dec_bf, cn, lmod, half);
      tile_wmma2(afrag, b1, a0, a1);
      upd1(a0, a1, b_dec[(c + NWAVES) * 16 + lmod]);
    }
    if (c < NCHUNK) {                           // odd leftover: b0 == chunk c
      tile_wmma2(afrag, b0, a0, a1);
      upd1(a0, a1, b_dec[c * 16 + lmod]);
    }
  }

  // lane-merge within each 16-lane half (rows do not cross halves)
#pragma unroll
  for (int off = 1; off < 16; off <<= 1) {
#pragma unroll
    for (int mt = 0; mt < MTILES; ++mt)
#pragma unroll
      for (int v = 0; v < 8; ++v) {
        const float om = __shfl_xor(m_run[mt][v], off, 32);
        const float os = __shfl_xor(s_run[mt][v], off, 32);
        const float mn = fmaxf(m_run[mt][v], om);
        s_run[mt][v] = s_run[mt][v] * __expf(m_run[mt][v] - mn) + os * __expf(om - mn);
        m_run[mt][v] = mn;
      }
  }
  if (lmod == 0) {
#pragma unroll
    for (int mt = 0; mt < MTILES; ++mt)
#pragma unroll
      for (int v = 0; v < 8; ++v) {
        wred_m[wave][mt * 16 + half * 8 + v] = m_run[mt][v];
        wred_s[wave][mt * 16 + half * 8 + v] = s_run[mt][v];
      }
  }
  __syncthreads();
  if (tid < ROWS) {
    float m = wred_m[0][tid], s = wred_s[0][tid];
#pragma unroll
    for (int w = 1; w < NWAVES; ++w) {
      const float om = wred_m[w][tid], os = wred_s[w][tid];
      const float mn = fmaxf(m, om);
      s = s * __expf(m - mn) + os * __expf(om - mn);
      m = mn;
    }
    M1[tid] = m; S1[tid] = s; PINV[tid] = 1.0f / s;  // PINV == max(probs)
  }
  __syncthreads();

  // ======================= PASS 2: S2 = sum exp(p - pmax) ==================
  float row_m[MTILES][8], row_pi[MTILES][8], s2[MTILES][8];
#pragma unroll
  for (int mt = 0; mt < MTILES; ++mt)
#pragma unroll
    for (int v = 0; v < 8; ++v) {
      const int r = mt * 16 + v + half * 8;
      row_m[mt][v] = M1[r]; row_pi[mt][v] = PINV[r]; s2[mt][v] = 0.0f;
    }

  auto upd2 = [&](const v8f& a0, const v8f& a1, float bd) {
#pragma unroll
    for (int v = 0; v < 8; ++v) {
      {
        const float p = __expf(a0[v] + bd - row_m[0][v]) * row_pi[0][v];
        s2[0][v] += __expf(p - row_pi[0][v]);   // pmax == row_pi
      }
      {
        const float p = __expf(a1[v] + bd - row_m[1][v]) * row_pi[1][v];
        s2[1][v] += __expf(p - row_pi[1][v]);
      }
    }
  };

  {
    int c = wave;
    BFrag b0 = load_chunk<PRECVT>(W_dec, W_dec_bf, c, lmod, half);
    v8f a0, a1;
    for (; c + NWAVES < NCHUNK; c += 2 * NWAVES) {
      BFrag b1 = load_chunk<PRECVT>(W_dec, W_dec_bf, c + NWAVES, lmod, half);
      tile_wmma2(afrag, b0, a0, a1);
      upd2(a0, a1, b_dec[c * 16 + lmod]);
      const int cn = (c + 2 * NWAVES < NCHUNK) ? (c + 2 * NWAVES) : (NCHUNK - 1);
      b0 = load_chunk<PRECVT>(W_dec, W_dec_bf, cn, lmod, half);
      tile_wmma2(afrag, b1, a0, a1);
      upd2(a0, a1, b_dec[(c + NWAVES) * 16 + lmod]);
    }
    if (c < NCHUNK) {
      tile_wmma2(afrag, b0, a0, a1);
      upd2(a0, a1, b_dec[c * 16 + lmod]);
    }
  }

#pragma unroll
  for (int off = 1; off < 16; off <<= 1) {
#pragma unroll
    for (int mt = 0; mt < MTILES; ++mt)
#pragma unroll
      for (int v = 0; v < 8; ++v) s2[mt][v] += __shfl_xor(s2[mt][v], off, 32);
  }
  if (lmod == 0) {
#pragma unroll
    for (int mt = 0; mt < MTILES; ++mt)
#pragma unroll
      for (int v = 0; v < 8; ++v)
        wred_s[wave][mt * 16 + half * 8 + v] = s2[mt][v];
  }
  __syncthreads();

  // ============================ Finalize rows ==============================
  if (tid < ROWS) {
    float t = 0.0f;
#pragma unroll
    for (int w = 0; w < NWAVES; ++w) t += wred_s[w][tid];
    const float pmax = PINV[tid];
    const float lse2 = pmax + __logf(t);
    const float ptgt = __expf(TLOGIT[tid] - M1[tid]) * PINV[tid];
    const float nll  = lse2 - ptgt;
    atomicAdd(out, nll * (1.0f / (float)T_TOTAL));
  }
}

// ---------------------------------------------------------------------------
extern "C" void kernel_launch(void* const* d_in, const int* in_sizes, int n_in,
                              void* d_out, int out_size, void* d_ws, size_t ws_size,
                              hipStream_t stream) {
  const float* W_enc  = (const float*)d_in[0];
  const float* b_enc  = (const float*)d_in[1];
  const float* W_dec  = (const float*)d_in[2];
  const float* b_dec  = (const float*)d_in[3];
  const int*   target = (const int*)d_in[4];
  const int*   rw     = (const int*)d_in[5];
  float* out = (float*)d_out;

  zero_out_kernel<<<1, 1, 0, stream>>>(out);

  const size_t bf_bytes = (size_t)NUM_NODES * ENC_DIM * sizeof(__bf16);
  if (ws_size >= bf_bytes) {
    __bf16* wbf = (__bf16*)d_ws;
    cvt_wdec_kernel<<<1024, 256, 0, stream>>>(W_dec, wbf, NUM_NODES * ENC_DIM);
    skipgram_loss_kernel<true><<<T_TOTAL / ROWS, NTHREADS, 0, stream>>>(
        W_enc, b_enc, W_dec, wbf, b_dec, target, rw, out);
  } else {
    skipgram_loss_kernel<false><<<T_TOTAL / ROWS, NTHREADS, 0, stream>>>(
        W_enc, b_enc, W_dec, nullptr, b_dec, target, rw, out);
  }
}
